// MVoTTokenProcessor_62972810494576
// MI455X (gfx1250) — matrix operand order repaired and checked by reading, not compiled, not run on hardware
//
#include <hip/hip_runtime.h>

// ---------------------------------------------------------------- constants
#define H     2048
#define NTOK  16384                 // B*S = 4*4096
#define HH    ((size_t)H * (size_t)H)

#define BM 128
#define BN 256
#define BK 32
#define SA 40      // LDS row stride (elements): 80B = 5*16B, conflict-free column access

typedef __attribute__((ext_vector_type(16))) __bf16 v16bf;
typedef __attribute__((ext_vector_type(8)))  float  v8f;

union FragU { v16bf v; unsigned int d[8]; };

__device__ __forceinline__ unsigned short f2bf(float f) {
    union { float f; unsigned int u; } a; a.f = f;
    unsigned int u = a.u;
    u += 0x7FFFu + ((u >> 16) & 1u);      // round-to-nearest-even
    return (unsigned short)(u >> 16);
}

// ------------------------------------------- weight f32 -> bf16 + transpose
// Input  W  [k][n] (f32, row-major, H x H)
// Output WT [n][k] (bf16) so GEMM B-tiles have contiguous k per n-row.
#define TDIM 32
__global__ void __launch_bounds__(256)
conv_wt(const float* __restrict__ s0, const float* __restrict__ s1,
        const float* __restrict__ s2, const float* __restrict__ s3,
        unsigned short* __restrict__ wb) {
    __shared__ __align__(8) unsigned short tile[TDIM][TDIM + 4]; // stride 36 elems = 72B (8B aligned)
    const int mat = blockIdx.z;
    const float* s = (mat == 0) ? s0 : (mat == 1) ? s1 : (mat == 2) ? s2 : s3;
    unsigned short* d = wb + (size_t)mat * HH;
    const int k0 = blockIdx.y * TDIM;   // source row block (k)
    const int n0 = blockIdx.x * TDIM;   // source col block (n)
    const int t  = threadIdx.x;

    // read 32x32 f32 coalesced: thread -> row r = t>>3, cols c..c+3
    const int r = t >> 3, c = (t & 7) * 4;
    float4 v = *(const float4*)(s + (size_t)(k0 + r) * H + n0 + c);
    tile[c + 0][r] = f2bf(v.x);
    tile[c + 1][r] = f2bf(v.y);
    tile[c + 2][r] = f2bf(v.z);
    tile[c + 3][r] = f2bf(v.w);
    __syncthreads();

    // write transposed: thread -> n-row nn = t>>3, k cols kc..kc+3 (one 8B store)
    const int nn = t >> 3, kc = (t & 7) * 4;
    unsigned int lo = (unsigned int)tile[nn][kc]     | ((unsigned int)tile[nn][kc + 1] << 16);
    unsigned int hi = (unsigned int)tile[nn][kc + 2] | ((unsigned int)tile[nn][kc + 3] << 16);
    uint2 pk = {lo, hi};
    *(uint2*)(d + (size_t)(n0 + nn) * H + k0 + kc) = pk;
}

// ---------------------------------------------------------------- LayerNorm + routing
__global__ void __launch_bounds__(256)
ln_route(const float* __restrict__ hs, const int* __restrict__ tt,
         const float* __restrict__ gamma, const float* __restrict__ beta,
         unsigned short* __restrict__ xg, int* __restrict__ perm,
         int* __restrict__ cnt) {
    const int tok = blockIdx.x;
    const int t   = threadIdx.x;
    const float* row = hs + (size_t)tok * H;

    __shared__ float2 red[256];
    __shared__ float  s_mu, s_rstd;
    __shared__ int    s_row;

    float vals[8];
    float sum = 0.f, sq = 0.f;
    #pragma unroll
    for (int i = 0; i < 2; ++i) {
        float4 x4 = ((const float4*)row)[t * 2 + i];
        vals[i*4+0] = x4.x; vals[i*4+1] = x4.y; vals[i*4+2] = x4.z; vals[i*4+3] = x4.w;
        sum += x4.x + x4.y + x4.z + x4.w;
        sq  += x4.x*x4.x + x4.y*x4.y + x4.z*x4.z + x4.w*x4.w;
    }
    red[t].x = sum; red[t].y = sq;
    __syncthreads();
    for (int s = 128; s > 0; s >>= 1) {
        if (t < s) { red[t].x += red[t+s].x; red[t].y += red[t+s].y; }
        __syncthreads();
    }
    if (t == 0) {
        float mu  = red[0].x * (1.0f / H);
        float var = red[0].y * (1.0f / H) - mu * mu;
        s_mu = mu; s_rstd = rsqrtf(var + 1e-12f);
        int ty  = tt[tok];
        int pos = atomicAdd(&cnt[ty], 1);
        int r   = (ty == 0) ? pos : (NTOK - 1 - pos);   // text from front, image from back
        s_row = r;
        perm[r] = tok;
    }
    __syncthreads();
    const float mu = s_mu, rs = s_rstd;
    unsigned short* dst = xg + (size_t)s_row * H + t * 8;
    unsigned int p[4];
    #pragma unroll
    for (int i = 0; i < 4; ++i) {
        int c0 = t * 8 + i * 2;
        float y0 = (vals[i*2+0] - mu) * rs * gamma[c0]     + beta[c0];
        float y1 = (vals[i*2+1] - mu) * rs * gamma[c0 + 1] + beta[c0 + 1];
        p[i] = (unsigned int)f2bf(y0) | ((unsigned int)f2bf(y1) << 16);
    }
    uint4 pk = {p[0], p[1], p[2], p[3]};
    *(uint4*)dst = pk;
}

// ---------------------------------------------------------------- WMMA fragment loads
// 16-bit A 16x32: lane m = lane&15; lanes 0-15 K pairs {0..7,16..23}; lanes 16-31 -> +8
__device__ __forceinline__ v16bf loadA(const unsigned short* As, int mBase, int lane) {
    FragU f;
    const unsigned short* p = As + (mBase + (lane & 15)) * SA + 8 * ((lane >> 4) & 1);
    #pragma unroll
    for (int v = 0; v < 4; ++v) f.d[v] = *(const unsigned int*)(p + 2 * v);
    #pragma unroll
    for (int v = 4; v < 8; ++v) f.d[v] = *(const unsigned int*)(p + 16 + 2 * (v - 4));
    return f.v;
}
// 16-bit B 32x16: lane n = lane&15; lanes 0-15 K pairs {0..7,16..23}; lanes 16-31 -> +8
// Bs is n-major (transposed weights), so each group of 4 dwords is contiguous.
__device__ __forceinline__ v16bf loadBt(const unsigned short* Bs, int nBase, int lane) {
    FragU f;
    const unsigned short* p = Bs + (nBase + (lane & 15)) * SA + 8 * ((lane >> 4) & 1);
    #pragma unroll
    for (int v = 0; v < 4; ++v) f.d[v] = *(const unsigned int*)(p + 2 * v);
    #pragma unroll
    for (int v = 4; v < 8; ++v) f.d[v] = *(const unsigned int*)(p + 16 + 2 * (v - 4));
    return f.v;
}

// ---------------------------------------------------------------- shared GEMM core
// C[BM x BN] += A[BM x H] * W[H x BN]; A row-major bf16, W given transposed (WT[n][k]) bf16.
// 8 waves as 2(M) x 4(N); each wave owns a 64x64 patch = 4x4 subtiles of 16x16.
// Register-staged pipeline: next global tile is loaded while WMMAs consume current LDS tile.
__device__ __forceinline__ void gemm_core(const unsigned short* __restrict__ Aglob,
                                          const unsigned short* __restrict__ WTglob,
                                          int m0, int n0,
                                          unsigned short* As, unsigned short* Bs,
                                          v8f acc[4][4]) {
    const int t     = threadIdx.x;
    const int lane  = t & 31;
    const int wid   = t >> 5;
    const int waveM = (wid >> 2) * 64;   // 2 waves along M (64 rows each)
    const int waveN = (wid & 3)  * 64;   // 4 waves along N (64 cols each)

    const int ar = t >> 1, ac = (t & 1) * 16;              // A tile: 128 rows x 32 k
    const unsigned short* aptr = Aglob  + (size_t)(m0 + ar) * H + ac;
    const unsigned short* bptr = WTglob + (size_t)(n0 + t) * H;   // B tile: 256 n-rows x 32 k, 1 row/thread

    uint4 a0 = ((const uint4*)aptr)[0], a1 = ((const uint4*)aptr)[1];
    uint4 b0 = ((const uint4*)bptr)[0], b1 = ((const uint4*)bptr)[1];
    uint4 b2 = ((const uint4*)bptr)[2], b3 = ((const uint4*)bptr)[3];

    uint4* adst = (uint4*)(As + ar * SA + ac);
    uint4* bdst = (uint4*)(Bs + t * SA);

    for (int kk = 0; kk < H; kk += BK) {
        adst[0] = a0; adst[1] = a1;
        bdst[0] = b0; bdst[1] = b1; bdst[2] = b2; bdst[3] = b3;
        __syncthreads();

        if (kk + BK < H) {                       // issue next tile's global loads early
            const unsigned short* an = aptr + kk + BK;
            const unsigned short* bn = bptr + kk + BK;
            a0 = ((const uint4*)an)[0]; a1 = ((const uint4*)an)[1];
            b0 = ((const uint4*)bn)[0]; b1 = ((const uint4*)bn)[1];
            b2 = ((const uint4*)bn)[2]; b3 = ((const uint4*)bn)[3];
        }

        v16bf bF[4];
        #pragma unroll
        for (int j = 0; j < 4; ++j) bF[j] = loadBt(Bs, waveN + j * 16, lane);
        #pragma unroll
        for (int i = 0; i < 4; ++i) {
            v16bf aF = loadA(As, waveM + i * 16, lane);
            #pragma unroll
            for (int j = 0; j < 4; ++j)
                acc[i][j] = __builtin_amdgcn_wmma_f32_16x16x32_bf16(
                    false, aF, false, bF[j], (short)0, acc[i][j], false, false);
        }
        __syncthreads();
    }
}

// ---------------------------------------------------------------- GEMM1: h = gelu(x @ W1 + b1)
__global__ void __launch_bounds__(256)
gemm1_kernel(const unsigned short* __restrict__ xg,
             const unsigned short* __restrict__ wb,
             const float* __restrict__ tb1, const float* __restrict__ ib1,
             unsigned short* __restrict__ hg,
             const int* __restrict__ cnt) {
    __shared__ __align__(16) unsigned short As[BM * SA];
    __shared__ __align__(16) unsigned short Bs[BN * SA];

    const int expert = blockIdx.z;
    const int count0 = cnt[0];
    const int m0 = blockIdx.y * BM, n0 = blockIdx.x * BN;
    if (expert == 0) { if (m0 >= count0)      return; }   // text rows: [0, count0)
    else             { if (m0 + BM <= count0) return; }   // image rows: [count0, NTOK)

    const unsigned short* WT = wb + (size_t)expert * 2 * HH;   // w1_text^T @0, w1_image^T @2*HH
    const float* bias = expert ? ib1 : tb1;

    v8f acc[4][4];
    const v8f zero = {0.f,0.f,0.f,0.f,0.f,0.f,0.f,0.f};
    #pragma unroll
    for (int i = 0; i < 4; ++i)
        #pragma unroll
        for (int j = 0; j < 4; ++j) acc[i][j] = zero;

    gemm_core(xg, WT, m0, n0, As, Bs, acc);

    const int lane = threadIdx.x & 31, wid = threadIdx.x >> 5;
    const int waveM = (wid >> 2) * 64, waveN = (wid & 3) * 64;
    const int lo = expert ? count0 : 0;
    const int hi = expert ? NTOK   : count0;
    #pragma unroll
    for (int i = 0; i < 4; ++i) {
        const int mB = m0 + waveM + i * 16 + ((lane >> 4) << 3);
        #pragma unroll
        for (int j = 0; j < 4; ++j) {
            const int col = n0 + waveN + j * 16 + (lane & 15);
            const float bv = bias[col];
            #pragma unroll
            for (int r = 0; r < 8; ++r) {
                const int row = mB + r;
                if (row >= lo && row < hi) {
                    float x = acc[i][j][r] + bv;
                    float g = 0.5f * x * (1.0f + erff(x * 0.70710678118654752f));
                    hg[(size_t)row * H + col] = f2bf(g);
                }
            }
        }
    }
}

// ------------------------------------------------ GEMM2: out[tok] = hidden[tok] + h @ W2 + b2
__global__ void __launch_bounds__(256)
gemm2_kernel(const unsigned short* __restrict__ hg,
             const unsigned short* __restrict__ wb,
             const float* __restrict__ tb2, const float* __restrict__ ib2,
             const int* __restrict__ perm,
             const float* __restrict__ hidden,
             float* __restrict__ out,
             const int* __restrict__ cnt) {
    __shared__ __align__(16) unsigned short As[BM * SA];
    __shared__ __align__(16) unsigned short Bs[BN * SA];

    const int expert = blockIdx.z;
    const int count0 = cnt[0];
    const int m0 = blockIdx.y * BM, n0 = blockIdx.x * BN;
    if (expert == 0) { if (m0 >= count0)      return; }
    else             { if (m0 + BM <= count0) return; }

    const unsigned short* WT = wb + HH + (size_t)expert * 2 * HH; // w2_text^T @HH, w2_image^T @3*HH
    const float* bias = expert ? ib2 : tb2;

    v8f acc[4][4];
    const v8f zero = {0.f,0.f,0.f,0.f,0.f,0.f,0.f,0.f};
    #pragma unroll
    for (int i = 0; i < 4; ++i)
        #pragma unroll
        for (int j = 0; j < 4; ++j) acc[i][j] = zero;

    gemm_core(hg, WT, m0, n0, As, Bs, acc);

    const int lane = threadIdx.x & 31, wid = threadIdx.x >> 5;
    const int waveM = (wid >> 2) * 64, waveN = (wid & 3) * 64;
    const int lo = expert ? count0 : 0;
    const int hi = expert ? NTOK   : count0;
    #pragma unroll
    for (int i = 0; i < 4; ++i) {
        const int mB = m0 + waveM + i * 16 + ((lane >> 4) << 3);
        #pragma unroll
        for (int j = 0; j < 4; ++j) {
            const int col = n0 + waveN + j * 16 + (lane & 15);
            const float bv = bias[col];
            #pragma unroll
            for (int r = 0; r < 8; ++r) {
                const int row = mB + r;
                if (row >= lo && row < hi) {
                    const int tok = perm[row];
                    const size_t o = (size_t)tok * H + col;
                    out[o] = hidden[o] + acc[i][j][r] + bv;
                }
            }
        }
    }
}

// ---------------------------------------------------------------- launcher
extern "C" void kernel_launch(void* const* d_in, const int* in_sizes, int n_in,
                              void* d_out, int out_size, void* d_ws, size_t ws_size,
                              hipStream_t stream) {
    const float* hidden = (const float*)d_in[0];
    const int*   tt     = (const int*)  d_in[1];
    const float* gamma  = (const float*)d_in[2];
    const float* beta   = (const float*)d_in[3];
    const float* tw1    = (const float*)d_in[4];
    const float* tb1    = (const float*)d_in[5];
    const float* tw2    = (const float*)d_in[6];
    const float* tb2    = (const float*)d_in[7];
    const float* iw1    = (const float*)d_in[8];
    const float* ib1    = (const float*)d_in[9];
    const float* iw2    = (const float*)d_in[10];
    const float* ib2    = (const float*)d_in[11];

    char* ws = (char*)d_ws;
    unsigned short* xg   = (unsigned short*)ws;                               // 64 MB gathered LN(x), bf16
    unsigned short* hg   = (unsigned short*)(ws + (size_t)NTOK * H * 2);      // 64 MB intermediate, bf16
    unsigned short* wb   = (unsigned short*)(ws + (size_t)2 * NTOK * H * 2);  // 32 MB bf16 W^T (w1t,w2t,w1i,w2i)
    int*            perm = (int*)(ws + (size_t)2 * NTOK * H * 2 + 4 * HH * 2);
    int*            cnt  = (int*)((char*)perm + (size_t)NTOK * sizeof(int));

    hipMemsetAsync(cnt, 0, 2 * sizeof(int), stream);

    // weights f32 -> bf16 transposed: wb order: [0]=tw1^T, [1]=tw2^T, [2]=iw1^T, [3]=iw2^T
    conv_wt<<<dim3(H / TDIM, H / TDIM, 4), 256, 0, stream>>>(tw1, tw2, iw1, iw2, wb);

    // LayerNorm + compaction routing
    ln_route<<<NTOK, 256, 0, stream>>>(hidden, tt, gamma, beta, xg, perm, cnt);

    // Routed MLP, both experts as grid.z
    dim3 grid(H / BN, NTOK / BM, 2);
    gemm1_kernel<<<grid, 256, 0, stream>>>(xg, wb, tb1, ib1, hg, cnt);
    gemm2_kernel<<<grid, 256, 0, stream>>>(hg, wb, tb2, ib2, perm, hidden, (float*)d_out, cnt);
}
